// BitLinear_24670292148852
// MI455X (gfx1250) — compile-verified, matrix operand order checked by reading, not run on hardware
//
#include <hip/hip_runtime.h>
#include <stdint.h>

#define EPS 1e-8f

// Problem constants (from reference): B*T = 4096 tokens, D_IN = 4096, D_OUT = 16384
#define K_DIM   4096
#define N_DIM   16384
#define W_ELEMS (67108864)   // 16384*4096
#define W_VEC4  (W_ELEMS/4)

typedef int v8i __attribute__((ext_vector_type(8)));
typedef int v4i_b128 __attribute__((vector_size(16)));   // matches builtin's param type

// ---------------------------------------------------------------------------
// Async global->LDS copy (CDNA5: GLOBAL_LOAD_ASYNC_TO_LDS_B128, ASYNCcnt).
// Signature (from clang diagnostic): (v4i AS1*, v4i AS3*, imm offset, imm cpol)
// ---------------------------------------------------------------------------
#if __has_builtin(__builtin_amdgcn_global_load_async_to_lds_b128)
#define ASYNC_COPY_B128(g, l)                                                      \
    __builtin_amdgcn_global_load_async_to_lds_b128(                                \
        (__attribute__((address_space(1))) v4i_b128*)(uintptr_t)(g),               \
        (__attribute__((address_space(3))) v4i_b128*)(unsigned)(uintptr_t)(l),     \
        0, 0)
#else
#define ASYNC_COPY_B128(g, l)                                                      \
    asm volatile("global_load_async_to_lds_b128 %0, %1, off"                       \
                 :: "v"((unsigned)(uintptr_t)(l)),                                 \
                    "v"((unsigned long long)(uintptr_t)(g))                        \
                 : "memory")
#endif

#if __has_builtin(__builtin_amdgcn_s_wait_asynccnt)
#define WAIT_ASYNC0() __builtin_amdgcn_s_wait_asynccnt(0)
#else
#define WAIT_ASYNC0() asm volatile("s_wait_asynccnt 0x0" ::: "memory")
#endif

// ---------------------------------------------------------------------------
// Pass 1a: per-block partial sums of |W|  (deterministic tree reduction)
// ---------------------------------------------------------------------------
__global__ __launch_bounds__(256) void wabs_partial(const float4* __restrict__ w4,
                                                    float* __restrict__ partial)
{
    int tid    = blockIdx.x * blockDim.x + threadIdx.x;
    int stride = gridDim.x * blockDim.x;
    float s = 0.0f;
    for (int i = tid; i < W_VEC4; i += stride) {
        float4 v = w4[i];
        s += fabsf(v.x) + fabsf(v.y) + fabsf(v.z) + fabsf(v.w);
    }
    __shared__ float red[256];
    red[threadIdx.x] = s;
    __syncthreads();
    for (int off = 128; off > 0; off >>= 1) {
        if (threadIdx.x < off) red[threadIdx.x] += red[threadIdx.x + off];
        __syncthreads();
    }
    if (threadIdx.x == 0) partial[blockIdx.x] = red[0];
}

// ---------------------------------------------------------------------------
// Pass 1b: final reduce -> scale_w = mean(|W|)
// ---------------------------------------------------------------------------
__global__ __launch_bounds__(256) void wabs_final(const float* __restrict__ partial,
                                                  float* __restrict__ scale_w, int n)
{
    float s = 0.0f;
    for (int i = threadIdx.x; i < n; i += 256) s += partial[i];
    __shared__ float red[256];
    red[threadIdx.x] = s;
    __syncthreads();
    for (int off = 128; off > 0; off >>= 1) {
        if (threadIdx.x < off) red[threadIdx.x] += red[threadIdx.x + off];
        __syncthreads();
    }
    if (threadIdx.x == 0) scale_w[0] = red[0] / (float)W_ELEMS;
}

// ---------------------------------------------------------------------------
// Pass 2: ternary-quantize weights to int8 {-1,0,1}, row-major (D_OUT x D_IN)
// ---------------------------------------------------------------------------
__global__ __launch_bounds__(256) void wquant(const float4* __restrict__ w4,
                                              const float* __restrict__ scale_w,
                                              char4* __restrict__ wq4)
{
    const float inv = 1.0f / (scale_w[0] + EPS);
    int tid    = blockIdx.x * blockDim.x + threadIdx.x;
    int stride = gridDim.x * blockDim.x;
    for (int i = tid; i < W_VEC4; i += stride) {
        float4 v = w4[i];
        char4 q;
        q.x = (signed char)fminf(1.0f, fmaxf(-1.0f, rintf(v.x * inv)));
        q.y = (signed char)fminf(1.0f, fmaxf(-1.0f, rintf(v.y * inv)));
        q.z = (signed char)fminf(1.0f, fmaxf(-1.0f, rintf(v.z * inv)));
        q.w = (signed char)fminf(1.0f, fmaxf(-1.0f, rintf(v.w * inv)));
        wq4[i] = q;
    }
}

// ---------------------------------------------------------------------------
// Pass 3: per-token int8 activation quant.  One block per token (K=4096).
// ---------------------------------------------------------------------------
__global__ __launch_bounds__(256) void xquant(const float* __restrict__ x,
                                              signed char* __restrict__ xq,
                                              float* __restrict__ scale_x)
{
    const int token = blockIdx.x;
    const float4* row = (const float4*)(x + (size_t)token * K_DIM);

    float m = 0.0f;
    for (int i = threadIdx.x; i < K_DIM / 4; i += 256) {
        float4 v = row[i];
        m = fmaxf(m, fmaxf(fmaxf(fabsf(v.x), fabsf(v.y)), fmaxf(fabsf(v.z), fabsf(v.w))));
    }
    __shared__ float red[256];
    red[threadIdx.x] = m;
    __syncthreads();
    for (int off = 128; off > 0; off >>= 1) {
        if (threadIdx.x < off) red[threadIdx.x] = fmaxf(red[threadIdx.x], red[threadIdx.x + off]);
        __syncthreads();
    }
    __shared__ float s_inv;
    if (threadIdx.x == 0) {
        float gamma = red[0];
        scale_x[token] = gamma / 127.0f;
        s_inv = 127.0f / (gamma + EPS);
    }
    __syncthreads();
    const float inv = s_inv;
    char4* q4 = (char4*)(xq + (size_t)token * K_DIM);
    for (int i = threadIdx.x; i < K_DIM / 4; i += 256) {
        float4 v = row[i];
        char4 q;
        q.x = (signed char)fminf(127.0f, fmaxf(-128.0f, rintf(v.x * inv)));
        q.y = (signed char)fminf(127.0f, fmaxf(-128.0f, rintf(v.y * inv)));
        q.z = (signed char)fminf(127.0f, fmaxf(-128.0f, rintf(v.z * inv)));
        q.w = (signed char)fminf(127.0f, fmaxf(-128.0f, rintf(v.w * inv)));
        q4[i] = q;
    }
}

// ---------------------------------------------------------------------------
// Pass 4: INT8 WMMA GEMM: out[m,n] = (sum_k xq[m,k]*wq[n,k]) * scale_w * scale_x[m]
// Block tile 128(M) x 128(N), K stepped by 64.
// Double-buffered LDS + async global->LDS DMA (ASYNCcnt), ONE barrier per K-step:
//   prologue: async-load tile 0 into buf 0
//   iter k  : s_wait_asynccnt 0 ; barrier ; async-load tile k+1 into buf^1 ;
//             WMMA on buf
// Safe because each wave's fragment ds_loads complete (DScnt) before its WMMAs
// consume them, which precedes its barrier arrival -> after the barrier, the
// buffer being overwritten (buf^1, last computed at k-1) is no longer read.
//
// 8 waves: waveM = wv&3 (32-row band), waveN = wv>>2 (64-col band),
// each wave: 2(M) x 4(N) = 8 accumulators of v_wmma_i32_16x16x64_iu8.
//
// 8-bit fragment layout (ISA 7.12.2): lane&15 = row (A) / col (B);
// lane half (lane>>4) selects interleaved 8-byte K-groups.
// ---------------------------------------------------------------------------
__global__ __launch_bounds__(256) void bitlinear_gemm(const signed char* __restrict__ xq,
                                                      const signed char* __restrict__ wq,
                                                      const float* __restrict__ scale_x,
                                                      const float* __restrict__ scale_w,
                                                      float* __restrict__ out)
{
    __shared__ __align__(16) signed char sA[2][128 * 64];
    __shared__ __align__(16) signed char sB[2][128 * 64];
    __shared__ float sSX[128];

    const int tid   = threadIdx.x;
    const int lane  = tid & 31;
    const int wv    = tid >> 5;
    const int waveM = wv & 3;   // 0..3 : 32-row band
    const int waveN = wv >> 2;  // 0..1 : 64-col band

    const int mBase = blockIdx.y * 128;
    const int nBase = blockIdx.x * 128;

    if (tid < 128) sSX[tid] = scale_x[mBase + tid];
    const float sw = scale_w[0];

    const int halfsel = lane >> 4;       // which K-half this lane holds
    const int mrow0   = waveM * 32 + (lane & 15);
    const int brow0   = waveN * 64 + (lane & 15);

    v8i acc[2][4];
    const v8i vzero = {0, 0, 0, 0, 0, 0, 0, 0};
#pragma unroll
    for (int mi = 0; mi < 2; ++mi)
#pragma unroll
        for (int ni = 0; ni < 4; ++ni) acc[mi][ni] = vzero;

    // tile-loading geometry: 512 16B chunks per 128x64 tile, 2 chunks/thread
    const int ld_row0 = tid >> 2;        // chunk/4 for chunk = tid
    const int ld_cg0  = tid & 3;
    const int ld_row1 = (tid + 256) >> 2;
    const int ld_cg1  = (tid + 256) & 3;

    const size_t gA0 = (size_t)(mBase + ld_row0) * K_DIM + ld_cg0 * 16;
    const size_t gA1 = (size_t)(mBase + ld_row1) * K_DIM + ld_cg1 * 16;
    const size_t gB0 = (size_t)(nBase + ld_row0) * K_DIM + ld_cg0 * 16;
    const size_t gB1 = (size_t)(nBase + ld_row1) * K_DIM + ld_cg1 * 16;
    const int lA0 = ld_row0 * 64 + ld_cg0 * 16;
    const int lA1 = ld_row1 * 64 + ld_cg1 * 16;

    auto issue_tile = [&](int buf, int kk) {
        ASYNC_COPY_B128(xq + gA0 + kk, &sA[buf][lA0]);
        ASYNC_COPY_B128(wq + gB0 + kk, &sB[buf][lA0]);
        ASYNC_COPY_B128(xq + gA1 + kk, &sA[buf][lA1]);
        ASYNC_COPY_B128(wq + gB1 + kk, &sB[buf][lA1]);
    };

    // prologue: stage K-tile 0 into buffer 0
    issue_tile(0, 0);

    for (int kk = 0; kk < K_DIM; kk += 64) {
        const int buf = (kk >> 6) & 1;

        WAIT_ASYNC0();       // this wave's async writes for tile kk landed
        __syncthreads();     // all waves' tile-kk data visible; old buffer free

        if (kk + 64 < K_DIM) issue_tile(buf ^ 1, kk + 64);

        // A fragments for this wave's two 16-row strips
        v8i afrag[2];
#pragma unroll
        for (int mi = 0; mi < 2; ++mi) {
            const signed char* base = &sA[buf][(mrow0 + mi * 16) * 64 + halfsel * 8];
            union { unsigned long long q[4]; v8i v; } u;
#pragma unroll
            for (int g = 0; g < 4; ++g) u.q[g] = *(const unsigned long long*)(base + g * 16);
            afrag[mi] = u.v;
        }
        // B fragments (one at a time) x both A strips
#pragma unroll
        for (int ni = 0; ni < 4; ++ni) {
            const signed char* base = &sB[buf][(brow0 + ni * 16) * 64 + halfsel * 8];
            union { unsigned long long q[4]; v8i v; } u;
#pragma unroll
            for (int g = 0; g < 4; ++g) u.q[g] = *(const unsigned long long*)(base + g * 16);
            const v8i bfrag = u.v;
#pragma unroll
            for (int mi = 0; mi < 2; ++mi) {
                // (sgn_a, A, sgn_b, B, C, reuse_a, reuse_b) - both operands signed int8
                acc[mi][ni] = __builtin_amdgcn_wmma_i32_16x16x64_iu8(
                    true, afrag[mi], true, bfrag, acc[mi][ni], false, false);
            }
        }
    }

    // Epilogue: dequant + store.  C layout: row = r + 8*(lane>=16), col = lane&15.
    const int colBase = nBase + waveN * 64 + (lane & 15);
#pragma unroll
    for (int mi = 0; mi < 2; ++mi) {
#pragma unroll
        for (int ni = 0; ni < 4; ++ni) {
#pragma unroll
            for (int r = 0; r < 8; ++r) {
                const int rloc = waveM * 32 + mi * 16 + r + halfsel * 8;
                const int col  = colBase + ni * 16;
                out[(size_t)(mBase + rloc) * N_DIM + col] =
                    (float)acc[mi][ni][r] * sw * sSX[rloc];
            }
        }
    }
}

// ---------------------------------------------------------------------------
// Host-side launcher
// ---------------------------------------------------------------------------
extern "C" void kernel_launch(void* const* d_in, const int* in_sizes, int n_in,
                              void* d_out, int out_size, void* d_ws, size_t ws_size,
                              hipStream_t stream)
{
    const float* x = (const float*)d_in[0];   // (B,T,D_IN) f32
    const float* w = (const float*)d_in[1];   // (D_OUT,D_IN) f32
    float* out = (float*)d_out;               // (B,T,D_OUT) f32

    const int M = in_sizes[0] / K_DIM;        // B*T tokens (4096)

    // workspace layout (256B-aligned regions)
    char* ws = (char*)d_ws;
    float*       partial  = (float*)(ws + 0);          // 4096 floats
    float*       scale_w  = (float*)(ws + 16384);      // 1 float
    float*       scale_x  = (float*)(ws + 16640);      // M floats
    signed char* wq       = (signed char*)(ws + 33280);            // 64 MB int8
    signed char* xq       = wq + (size_t)N_DIM * K_DIM;            // 16 MB int8

    (void)n_in; (void)out_size; (void)ws_size;

    wabs_partial<<<4096, 256, 0, stream>>>((const float4*)w, partial);
    wabs_final<<<1, 256, 0, stream>>>(partial, scale_w, 4096);
    wquant<<<8192, 256, 0, stream>>>((const float4*)w, scale_w, (char4*)wq);
    xquant<<<M, 256, 0, stream>>>(x, xq, scale_x);

    dim3 grid(N_DIM / 128, M / 128);   // (128, 32)
    bitlinear_gemm<<<grid, 256, 0, stream>>>(xq, wq, scale_x, scale_w, out);
}
